// PredictBlock_10660108829387
// MI455X (gfx1250) — matrix operand order repaired and checked by reading, not compiled, not run on hardware
//
#include <hip/hip_runtime.h>

#define H_  16
#define D_  32
#define C1_ 1024
#define C2_ 512
#define S_  2048
#define M_  512
#define B_  2
#define W_  4

typedef _Float16 half_t;
typedef __attribute__((ext_vector_type(16))) _Float16 v16h;
typedef __attribute__((ext_vector_type(8)))  _Float16 v8h;
typedef __attribute__((ext_vector_type(8)))  float    v8f;
typedef __attribute__((ext_vector_type(4)))  int      v4i;

// --------------------------- CDNA5 async-to-LDS ----------------------------
#if defined(__has_builtin)
#if __has_builtin(__builtin_amdgcn_global_load_async_to_lds_b128)
#define HAVE_ASYNC_LDS 1
#endif
#endif
#ifndef HAVE_ASYNC_LDS
#define HAVE_ASYNC_LDS 0
#endif

__device__ __forceinline__ void stage16B(const half_t* __restrict__ g, half_t* l) {
#if HAVE_ASYNC_LDS
  typedef __attribute__((address_space(1))) v4i* gp_t;
  typedef __attribute__((address_space(3))) v4i* lp_t;
  half_t* gnc = (half_t*)g;  // strip const, then addrspace-cast
  __builtin_amdgcn_global_load_async_to_lds_b128((gp_t)gnc, (lp_t)l, 0, 0);
#else
  *(v8h*)l = *(const v8h*)g;
#endif
}

// Wait until at most N async ops remain outstanding (loads retire in order,
// so N == one stage's per-wave op count keeps the next stage in flight).
template <int N>
__device__ __forceinline__ void wait_async() {
#if HAVE_ASYNC_LDS
#if defined(__has_builtin) && __has_builtin(__builtin_amdgcn_s_wait_asynccnt)
  __builtin_amdgcn_s_wait_asynccnt(N);
#else
  asm volatile("s_wait_asynccnt %0" :: "i"(N) : "memory");
#endif
#endif
}

// ---------------------------------------------------------------------------
// Fragment loaders (wave32 WMMA layouts per CDNA5 ISA 7.12.2)
// A 16x32 f16: lane(0..15)=row, elems 0-7 = K[ha*8..+7], elems 8-15 = K[16+ha*8..+7]
// B 32x16 f16 from Bt (NxK row-major): lane=col, elems 0-7 = K[ha*8..+7], elems 8-15 = K[16+ha*8..+7]
// ---------------------------------------------------------------------------
__device__ __forceinline__ v16h combine(v8h lo, v8h hi) {
  v16h f;
#pragma unroll
  for (int t = 0; t < 8; ++t) { f[t] = lo[t]; f[t + 8] = hi[t]; }
  return f;
}

__device__ __forceinline__ v16h load_a_frag(const half_t* __restrict__ base, long ld,
                                            int row, int k0, int ha) {
  const half_t* p = base + (long)row * ld + k0 + ha * 8;
  return combine(*(const v8h*)(p), *(const v8h*)(p + 16));
}

__device__ __forceinline__ v16h load_b_frag(const half_t* __restrict__ base, long ld,
                                            int row, int k0, int ha) {
  const half_t* p = base + (long)row * ld + k0 + ha * 8;
  return combine(*(const v8h*)(p), *(const v8h*)(p + 16));
}

// ---------------------------------------------------------------------------
// Guard-free WMMA GEMM, double-buffered async-to-LDS staging.
// Block = 256 threads = 8 waves arranged WMW x WNW; wave tile 32x32.
// Block tile (WMW*32) x (WNW*32). All launched shapes divide exactly.
// ---------------------------------------------------------------------------
template <int WMW, int WNW>
__global__ void __launch_bounds__(256) gemm_f16_kernel(
    const half_t* __restrict__ A, long lda, long strideA, int divA,
    const half_t* __restrict__ Bt, long ldb, long strideB, int modB,
    float* __restrict__ outF, half_t* __restrict__ outH, long ldo, long strideO,
    const float* __restrict__ bias,
    const float* __restrict__ res, const float* __restrict__ res2,
    int K, int act)
{
  constexpr int TM = WMW * 32;
  constexpr int TN = WNW * 32;
  constexpr int ACH = TM * 4;             // 16B chunks in A tile (4 per row)
  constexpr int BCH = TN * 4;             // 16B chunks in B tile
  constexpr int APT = ACH / 256;          // A chunks per thread (>=1)
  constexpr int BPT = BCH / 256;          // B chunks per thread (0 -> lane<16 path)
  constexpr int STAGE_OPS = APT + (BPT >= 1 ? BPT : 1);  // async ops/wave/stage

  __shared__ __align__(16) half_t As[2][TM * 32];
  __shared__ __align__(16) half_t Bs[2][TN * 32];

  const int z = blockIdx.z;
  const half_t* Az = A + (long)(z / divA) * strideA;
  const half_t* Bz = Bt + (long)(z % modB) * strideB;
  const long oOff = (long)z * strideO;

  const int tid = threadIdx.x;
  const int lane = tid & 31;
  const int wave = tid >> 5;
  const int ha = lane >> 4;
  const int lr = lane & 15;
  const int mBlk = blockIdx.y * TM;
  const int nBlk = blockIdx.x * TN;
  const int wm = (wave / WNW) * 32;   // wave row offset inside tile
  const int wn = (wave % WNW) * 32;   // wave col offset inside tile

  auto stage = [&](int pb, int k0) {
#pragma unroll
    for (int c = 0; c < APT; ++c) {
      const int ch = c * 256 + tid;
      const int row = ch >> 2, c8 = (ch & 3) * 8;
      stage16B(Az + (long)(mBlk + row) * lda + k0 + c8, &As[pb][row * 32 + c8]);
    }
    if (BPT >= 1) {
#pragma unroll
      for (int c = 0; c < (BPT >= 1 ? BPT : 1); ++c) {
        const int ch = c * 256 + tid;
        const int row = ch >> 2, c8 = (ch & 3) * 8;
        stage16B(Bz + (long)(nBlk + row) * ldb + k0 + c8, &Bs[pb][row * 32 + c8]);
      }
    } else {
      // BCH == 128: one chunk per wave-halflane so every wave issues exactly
      // one async op (keeps per-wave ASYNCcnt uniform for the partial wait)
      if (lane < 16) {
        const int ch = wave * 16 + lane;
        const int row = ch >> 2, c8 = (ch & 3) * 8;
        stage16B(Bz + (long)(nBlk + row) * ldb + k0 + c8, &Bs[pb][row * 32 + c8]);
      }
    }
  };

  v8f acc[2][2];
#pragma unroll
  for (int i = 0; i < 2; ++i)
#pragma unroll
    for (int j = 0; j < 2; ++j)
#pragma unroll
      for (int r = 0; r < 8; ++r) acc[i][j][r] = 0.0f;

  stage(0, 0);
  int pb = 0;
  for (int k0 = 0; k0 < K; k0 += 32, pb ^= 1) {
    if (k0 + 32 < K) {
      stage(pb ^ 1, k0 + 32);        // prefetch next stage into other buffer
      wait_async<STAGE_OPS>();       // retire previous stage only
    } else {
      wait_async<0>();               // last stage: drain
    }
    __syncthreads();

    // ---- fragments from LDS buffer pb
    v16h a[2], b[2];
#pragma unroll
    for (int i = 0; i < 2; ++i) {
      const half_t* p = &As[pb][(wm + 16 * i + lr) * 32 + ha * 8];
      a[i] = combine(*(const v8h*)p, *(const v8h*)(p + 16));
    }
#pragma unroll
    for (int j = 0; j < 2; ++j) {
      const half_t* p = &Bs[pb][(wn + 16 * j + lr) * 32 + ha * 8];
      b[j] = combine(*(const v8h*)p, *(const v8h*)(p + 16));
    }
#pragma unroll
    for (int i = 0; i < 2; ++i)
#pragma unroll
      for (int j = 0; j < 2; ++j)
        acc[i][j] = __builtin_amdgcn_wmma_f32_16x16x32_f16(
            false, a[i], false, b[j], (short)0, acc[i][j], false, false);
    __syncthreads();   // reads done before this buffer is restaged
  }

  // ---- epilogue: bias + residual(s) + optional exact GELU
#pragma unroll
  for (int i = 0; i < 2; ++i) {
#pragma unroll
    for (int j = 0; j < 2; ++j) {
      const int col = nBlk + wn + 16 * j + lr;
      const float bv = bias ? bias[col] : 0.0f;
#pragma unroll
      for (int r = 0; r < 8; ++r) {
        const int row = mBlk + wm + 16 * i + r + 8 * ha;
        float v = acc[i][j][r] + bv;
        const long ridx = (long)row * ldo + col;
        if (res)  v += res[ridx];
        if (res2) v += res2[ridx];
        if (act == 1) v = 0.5f * v * (1.0f + erff(v * 0.70710678118654752440f));
        const long idx = oOff + ridx;
        if (outF) outF[idx] = v;
        if (outH) outH[idx] = (half_t)v;
      }
    }
  }
}

// ---------------------------------------------------------------------------
// Flash-style attention. One wave per 32 query rows per (b,h).
// Post-softmax triu mask: denominator over ALL t, P-V only where t > s.
// q:[BH,S,D] k:[BH,T,D] vt:[BH,D,T]  out:[B,S,H*D] (head-major concat), f16.
// ---------------------------------------------------------------------------
__global__ void __launch_bounds__(32) attn_kernel(
    const half_t* __restrict__ q, const half_t* __restrict__ k,
    const half_t* __restrict__ vt, half_t* __restrict__ out,
    int S, int T, int causal)
{
  __shared__ __align__(16) half_t p_lds[32 * 32];
  const int z = blockIdx.y;
  const int bb = z / H_;
  const int hh = z % H_;
  const half_t* qz = q + (long)z * S * D_;
  const half_t* kz = k + (long)z * T * D_;
  const half_t* vz = vt + (long)z * D_ * T;
  half_t* oz = out + (long)bb * S * (H_ * D_) + hh * D_;

  const int lane = threadIdx.x;
  const int ha = lane >> 4;
  const int lr = lane & 15;
  const int s0 = blockIdx.x * 32;
  const float scale = 0.17677669529663688f; // 1/sqrt(32)

  v16h qa[2];
#pragma unroll
  for (int i = 0; i < 2; ++i) qa[i] = load_a_frag(qz, D_, s0 + 16 * i + lr, 0, ha);

  float m_run[2][8], l_run[2][8];
  v8f oacc[2][2];
#pragma unroll
  for (int i = 0; i < 2; ++i)
#pragma unroll
    for (int r = 0; r < 8; ++r) { m_run[i][r] = -3.0e38f; l_run[i][r] = 0.0f; }
#pragma unroll
  for (int i = 0; i < 2; ++i)
#pragma unroll
    for (int jd = 0; jd < 2; ++jd)
#pragma unroll
      for (int r = 0; r < 8; ++r) oacc[i][jd][r] = 0.0f;

  for (int t0 = 0; t0 < T; t0 += 32) {
    v16h kf[2];
#pragma unroll
    for (int j = 0; j < 2; ++j) kf[j] = load_b_frag(kz, D_, t0 + 16 * j + lr, 0, ha);

    v8f sc[2][2];
#pragma unroll
    for (int i = 0; i < 2; ++i)
#pragma unroll
      for (int j = 0; j < 2; ++j) {
#pragma unroll
        for (int r = 0; r < 8; ++r) sc[i][j][r] = 0.0f;
        sc[i][j] = __builtin_amdgcn_wmma_f32_16x16x32_f16(
            false, qa[i], false, kf[j], (short)0, sc[i][j], false, false);
      }

#pragma unroll
    for (int i = 0; i < 2; ++i) {
      float resc[8];
#pragma unroll
      for (int r = 0; r < 8; ++r) {
        const float a0 = sc[i][0][r] * scale;
        const float a1 = sc[i][1][r] * scale;
        float mx = fmaxf(a0, a1);
        mx = fmaxf(mx, __shfl_xor(mx, 1, 32));
        mx = fmaxf(mx, __shfl_xor(mx, 2, 32));
        mx = fmaxf(mx, __shfl_xor(mx, 4, 32));
        mx = fmaxf(mx, __shfl_xor(mx, 8, 32));
        const float nmax = fmaxf(m_run[i][r], mx);
        const float p0 = __expf(a0 - nmax);
        const float p1 = __expf(a1 - nmax);
        float ps = p0 + p1;
        ps += __shfl_xor(ps, 1, 32);
        ps += __shfl_xor(ps, 2, 32);
        ps += __shfl_xor(ps, 4, 32);
        ps += __shfl_xor(ps, 8, 32);
        const float rs = __expf(m_run[i][r] - nmax);
        l_run[i][r] = l_run[i][r] * rs + ps;
        m_run[i][r] = nmax;
        resc[r] = rs;
        const int row_l = 16 * i + r + 8 * ha;
        const int srow = s0 + row_l;
        // Mask applied only to the P used for P@V (denominator keeps full row)
        p_lds[row_l * 32 + lr]      = (half_t)((causal && (t0 + lr)      <= srow) ? 0.0f : p0);
        p_lds[row_l * 32 + 16 + lr] = (half_t)((causal && (t0 + 16 + lr) <= srow) ? 0.0f : p1);
      }
#pragma unroll
      for (int jd = 0; jd < 2; ++jd)
#pragma unroll
        for (int r = 0; r < 8; ++r) oacc[i][jd][r] *= resc[r];
    }
    __syncthreads();

    v16h pf[2], vf[2];
#pragma unroll
    for (int i = 0; i < 2; ++i) {  // A-layout read of P from LDS
      const half_t* pp = &p_lds[(16 * i + lr) * 32 + ha * 8];
      pf[i] = combine(*(const v8h*)pp, *(const v8h*)(pp + 16));
    }
#pragma unroll
    for (int jd = 0; jd < 2; ++jd) vf[jd] = load_b_frag(vz, T, 16 * jd + lr, t0, ha);
#pragma unroll
    for (int i = 0; i < 2; ++i)
#pragma unroll
      for (int jd = 0; jd < 2; ++jd)
        oacc[i][jd] = __builtin_amdgcn_wmma_f32_16x16x32_f16(
            false, pf[i], false, vf[jd], (short)0, oacc[i][jd], false, false);
    __syncthreads();
  }

#pragma unroll
  for (int i = 0; i < 2; ++i)
#pragma unroll
    for (int jd = 0; jd < 2; ++jd)
#pragma unroll
      for (int r = 0; r < 8; ++r) {
        const int row_l = 16 * i + r + 8 * ha;
        const float vv = oacc[i][jd][r] / l_run[i][r];
        oz[(long)(s0 + row_l) * (H_ * D_) + 16 * jd + lr] = (half_t)vv;
      }
}

// ---------------------------------------------------------------------------
// LayerNorm: one block per row. Optional f32 and f16 outputs.
// ---------------------------------------------------------------------------
__global__ void __launch_bounds__(256) layernorm_kernel(
    const float* __restrict__ x, const float* __restrict__ g, const float* __restrict__ b,
    float* __restrict__ outF, half_t* __restrict__ outH, int C)
{
  const long row = blockIdx.x;
  const float* xr = x + row * C;
  __shared__ float s1[256], s2[256];
  float sum = 0.f, sq = 0.f;
  for (int c = threadIdx.x; c < C; c += 256) { float v = xr[c]; sum += v; sq += v * v; }
  s1[threadIdx.x] = sum; s2[threadIdx.x] = sq;
  __syncthreads();
  for (int st = 128; st > 0; st >>= 1) {
    if ((int)threadIdx.x < st) {
      s1[threadIdx.x] += s1[threadIdx.x + st];
      s2[threadIdx.x] += s2[threadIdx.x + st];
    }
    __syncthreads();
  }
  const float mean = s1[0] / C;
  const float var = s2[0] / C - mean * mean;
  const float rstd = rsqrtf(var + 1e-5f);
  for (int c = threadIdx.x; c < C; c += 256) {
    const float v = (xr[c] - mean) * rstd * g[c] + b[c];
    if (outF) outF[row * C + c] = v;
    if (outH) outH[row * C + c] = (half_t)v;
  }
}

// ---------------------------------------------------------------------------
// Transpose + convert kernels (batched over blockIdx.y)
// ---------------------------------------------------------------------------
__global__ void __launch_bounds__(256) transpose_f32_f16_kernel(
    const float* __restrict__ src, half_t* __restrict__ dst, int R, int C)
{
  const long n = (long)R * C;
  const long idx = (long)blockIdx.x * 256 + threadIdx.x;
  if (idx >= n) return;
  const long zoff = (long)blockIdx.y * n;
  const int r = (int)(idx / C), c = (int)(idx % C);
  dst[zoff + (long)c * R + r] = (half_t)src[zoff + idx];
}

__global__ void __launch_bounds__(256) transpose_f16_kernel(
    const half_t* __restrict__ src, half_t* __restrict__ dst, int R, int C)
{
  const long n = (long)R * C;
  const long idx = (long)blockIdx.x * 256 + threadIdx.x;
  if (idx >= n) return;
  const long zoff = (long)blockIdx.y * n;
  const int r = (int)(idx / C), c = (int)(idx % C);
  dst[zoff + (long)c * R + r] = src[zoff + idx];
}

// ---------------------------------------------------------------------------
// Host side
// ---------------------------------------------------------------------------
struct MhaParams {
  const float *wq, *wk, *wv, *li1_w, *li1_b, *lnx_g, *lnx_b, *lny_g, *lny_b,
              *ln3_g, *ln3_b, *li2_w, *li2_b, *li3_w, *li3_b;
};

// cfg 0: 256x32 tile (WMW=8, WNW=1)  — QKV (N=32)
// cfg 1: 128x64 tile (WMW=4, WNW=2)  — wide GEMMs
static void launch_gemm(hipStream_t s, int cfg,
                        const half_t* A, long lda, long sA, int divA,
                        const half_t* Bt, long ldb, long sB, int modB,
                        float* oF, half_t* oH, long ldo, long sO,
                        const float* bias, const float* res, const float* res2,
                        int M, int N, int K, int Z, int act)
{
  if (cfg == 0) {
    dim3 g((unsigned)(N / 32), (unsigned)(M / 256), (unsigned)Z);
    gemm_f16_kernel<8, 1><<<g, dim3(256), 0, s>>>(A, lda, sA, divA, Bt, ldb, sB, modB,
                                                  oF, oH, ldo, sO, bias, res, res2, K, act);
  } else {
    dim3 g((unsigned)(N / 64), (unsigned)(M / 128), (unsigned)Z);
    gemm_f16_kernel<4, 2><<<g, dim3(256), 0, s>>>(A, lda, sA, divA, Bt, ldb, sB, modB,
                                                  oF, oH, ldo, sO, bias, res, res2, K, act);
  }
}

static void run_mha(hipStream_t st, const float* x, const float* y, int T, int cy,
                    const MhaParams& P, int causal,
                    const float* outer_res, float* final_out,
                    half_t* xn, half_t* yn, half_t* qb, half_t* kbuf, half_t* vbuf,
                    half_t* vtb, half_t* attn, float* xout, half_t* hn, half_t* hb,
                    half_t* wqt, half_t* wkt, half_t* wvt,
                    half_t* li1t, half_t* li2t, half_t* li3t)
{
  // Weight conversions / transposes -> f16 "Bt" layouts
  { long n = (long)C1_ * D_;
    transpose_f32_f16_kernel<<<dim3((unsigned)((n + 255) / 256), H_), 256, 0, st>>>(P.wq, wqt, C1_, D_); }
  { long n = (long)cy * D_;
    transpose_f32_f16_kernel<<<dim3((unsigned)((n + 255) / 256), H_), 256, 0, st>>>(P.wk, wkt, cy, D_);
    transpose_f32_f16_kernel<<<dim3((unsigned)((n + 255) / 256), H_), 256, 0, st>>>(P.wv, wvt, cy, D_); }
  { long n = (long)(H_ * D_) * C1_;
    transpose_f32_f16_kernel<<<dim3((unsigned)((n + 255) / 256), 1), 256, 0, st>>>(P.li1_w, li1t, H_ * D_, C1_); }
  { long n = (long)C1_ * (C1_ * W_);
    transpose_f32_f16_kernel<<<dim3((unsigned)((n + 255) / 256), 1), 256, 0, st>>>(P.li2_w, li2t, C1_, C1_ * W_);
    transpose_f32_f16_kernel<<<dim3((unsigned)((n + 255) / 256), 1), 256, 0, st>>>(P.li3_w, li3t, C1_ * W_, C1_); }

  // xn = LN(x; lnx), yn = LN(y; lny)
  layernorm_kernel<<<dim3(B_ * S_), dim3(256), 0, st>>>(x, P.lnx_g, P.lnx_b, nullptr, xn, C1_);
  layernorm_kernel<<<dim3((unsigned)(B_ * T)), dim3(256), 0, st>>>(y, P.lny_g, P.lny_b, nullptr, yn, cy);

  // q/k/v projections, batched over z = b*H + h   (N=32 -> cfg 0)
  launch_gemm(st, 0, xn, C1_, (long)S_ * C1_, H_, wqt, C1_, (long)D_ * C1_, H_,
              nullptr, qb, D_, (long)S_ * D_, nullptr, nullptr, nullptr,
              S_, D_, C1_, B_ * H_, 0);
  launch_gemm(st, 0, yn, cy, (long)T * cy, H_, wkt, cy, (long)D_ * cy, H_,
              nullptr, kbuf, D_, (long)T * D_, nullptr, nullptr, nullptr,
              T, D_, cy, B_ * H_, 0);
  launch_gemm(st, 0, yn, cy, (long)T * cy, H_, wvt, cy, (long)D_ * cy, H_,
              nullptr, vbuf, D_, (long)T * D_, nullptr, nullptr, nullptr,
              T, D_, cy, B_ * H_, 0);

  // vt[b,h] = v[b,h]^T  ([T,D] -> [D,T])
  { long n = (long)T * D_;
    transpose_f16_kernel<<<dim3((unsigned)((n + 255) / 256), (unsigned)(B_ * H_)), 256, 0, st>>>(vbuf, vtb, T, D_); }

  // attention -> attn [B,S,H*D] f16
  attn_kernel<<<dim3(S_ / 32, B_ * H_), dim3(32), 0, st>>>(qb, kbuf, vtb, attn, S_, T, causal);

  // x_out = x + attn @ li1_w + li1_b   (f32)
  launch_gemm(st, 1, attn, H_ * D_, 0, 1, li1t, H_ * D_, 0, 1,
              xout, nullptr, C1_, 0, P.li1_b, x, nullptr,
              B_ * S_, C1_, H_ * D_, 1, 0);

  // hn = LN(x_out; ln3)
  layernorm_kernel<<<dim3(B_ * S_), dim3(256), 0, st>>>(xout, P.ln3_g, P.ln3_b, nullptr, hn, C1_);

  // h = gelu(hn @ li2_w + li2_b)   (f16)
  launch_gemm(st, 1, hn, C1_, 0, 1, li2t, C1_, 0, 1,
              nullptr, hb, C1_ * W_, 0, P.li2_b, nullptr, nullptr,
              B_ * S_, C1_ * W_, C1_, 1, 1);

  // final = outer_res + x_out + h @ li3_w + li3_b
  launch_gemm(st, 1, hb, C1_ * W_, 0, 1, li3t, C1_ * W_, 0, 1,
              final_out, nullptr, C1_, 0, P.li3_b, xout, outer_res,
              B_ * S_, C1_, C1_ * W_, 1, 0);
}

extern "C" void kernel_launch(void* const* d_in, const int* in_sizes, int n_in,
                              void* d_out, int out_size, void* d_ws, size_t ws_size,
                              hipStream_t stream) {
  (void)in_sizes; (void)n_in; (void)out_size; (void)ws_size;
  const float* xx = (const float*)d_in[0];
  const float* ee = (const float*)d_in[1];
  const float* ln1_g = (const float*)d_in[2];
  const float* ln1_b = (const float*)d_in[3];
  const float* ln2_g = (const float*)d_in[4];
  const float* ln2_b = (const float*)d_in[5];
  MhaParams a1, a2;
  {
    const float** p = (const float**)&a1;
    for (int i = 0; i < 15; ++i) p[i] = (const float*)d_in[6 + i];
    const float** q = (const float**)&a2;
    for (int i = 0; i < 15; ++i) q[i] = (const float*)d_in[21 + i];
  }

  // workspace bump allocator
  char* w = (char*)d_ws;
  size_t off = 0;
  auto alloc = [&](size_t bytes) -> void* {
    void* p = (void*)(w + off);
    off += (bytes + 255) & ~(size_t)255;
    return p;
  };
  const size_t nBSC = (size_t)B_ * S_ * C1_;
  float*  x1   = (float*)alloc(nBSC * 4);
  float*  y1   = (float*)alloc(nBSC * 4);
  float*  x2   = (float*)alloc(nBSC * 4);
  float*  xout = (float*)alloc(nBSC * 4);
  half_t* xn   = (half_t*)alloc(nBSC * 2);
  half_t* yn   = (half_t*)alloc(nBSC * 2);
  half_t* qb   = (half_t*)alloc((size_t)B_ * H_ * S_ * D_ * 2);
  half_t* kbuf = (half_t*)alloc((size_t)B_ * H_ * S_ * D_ * 2);
  half_t* vbuf = (half_t*)alloc((size_t)B_ * H_ * S_ * D_ * 2);
  half_t* vtb  = (half_t*)alloc((size_t)B_ * H_ * S_ * D_ * 2);
  half_t* attn = (half_t*)alloc((size_t)B_ * S_ * H_ * D_ * 2);
  half_t* hn   = (half_t*)alloc(nBSC * 2);
  half_t* hb   = (half_t*)alloc(nBSC * W_ * 2);
  half_t* wqt  = (half_t*)alloc((size_t)H_ * D_ * C1_ * 2);
  half_t* wkt  = (half_t*)alloc((size_t)H_ * D_ * C1_ * 2);
  half_t* wvt  = (half_t*)alloc((size_t)H_ * D_ * C1_ * 2);
  half_t* li1t = (half_t*)alloc((size_t)C1_ * H_ * D_ * 2);
  half_t* li2t = (half_t*)alloc((size_t)C1_ * C1_ * W_ * 2);
  half_t* li3t = (half_t*)alloc((size_t)C1_ * C1_ * W_ * 2);

  // x1 = LN(xx; ln1)
  layernorm_kernel<<<dim3(B_ * S_), dim3(256), 0, stream>>>(xx, ln1_g, ln1_b, x1, nullptr, C1_);

  // y1 = x1 + mha1(x1, x1, causal)
  run_mha(stream, x1, x1, S_, C1_, a1, /*causal=*/1, x1, y1,
          xn, yn, qb, kbuf, vbuf, vtb, attn, xout, hn, hb,
          wqt, wkt, wvt, li1t, li2t, li3t);

  // x2 = LN(y1; ln2)
  layernorm_kernel<<<dim3(B_ * S_), dim3(256), 0, stream>>>(y1, ln2_g, ln2_b, x2, nullptr, C1_);

  // out = x2 + mha2(x2, ee, non-causal)
  run_mha(stream, x2, ee, M_, C2_, a2, /*causal=*/0, x2, (float*)d_out,
          xn, yn, qb, kbuf, vbuf, vtb, attn, xout, hn, hb,
          wqt, wkt, wvt, li1t, li2t, li3t);
}
